// LSTMSeq2seq_31293131719240
// MI455X (gfx1250) — matrix operand order repaired and checked by reading, not compiled, not run on hardware
//
#include <hip/hip_runtime.h>
#include <hip/hip_bf16.h>
#include <math.h>

// ---------------------------------------------------------------------------
// Problem constants (match reference)
// ---------------------------------------------------------------------------
constexpr int B_   = 64;
constexpr int TS_  = 64;
constexpr int TT_  = 64;
constexpr int E_   = 512;
constexpr int H_   = 512;      // per-direction encoder hidden
constexpr int DH_  = 1024;     // decoder hidden = 2H
constexpr int VT_  = 32000;
constexpr int FH_  = 4 * H_;   // 2048
constexpr int FDH_ = 4 * DH_;  // 4096

typedef __attribute__((ext_vector_type(16))) _Float16 v16h;
typedef __attribute__((ext_vector_type(8)))  _Float16 v8h;
typedef __attribute__((ext_vector_type(8)))  float    v8f;

static __device__ __forceinline__ float sigm(float x) { return 1.f / (1.f + __expf(-x)); }

// ---------------------------------------------------------------------------
// Utility kernels
// ---------------------------------------------------------------------------
__global__ void k_zero_f32(float* p, int n) {
  int i = blockIdx.x * blockDim.x + threadIdx.x;
  if (i < n) p[i] = 0.f;
}
__global__ void k_zero_f16(_Float16* p, int n) {
  int i = blockIdx.x * blockDim.x + threadIdx.x;
  if (i < n) p[i] = (_Float16)0.f;
}
__global__ void k_f32_to_f16(const float* __restrict__ s, _Float16* __restrict__ d, long n) {
  long i = (long)blockIdx.x * blockDim.x + threadIdx.x;
  if (i < n) d[i] = (_Float16)s[i];
}
// Embedding gather (rows of emb -> f16), n = rows * E
__global__ void k_gather_rows_f16(const int* __restrict__ tok, const float* __restrict__ emb,
                                  _Float16* __restrict__ out, int E, long n) {
  long i = (long)blockIdx.x * blockDim.x + threadIdx.x;
  if (i >= n) return;
  long row = i / E, col = i - row * E;
  out[i] = (_Float16)emb[(size_t)tok[row] * E + col];
}

// ---------------------------------------------------------------------------
// WMMA GEMM:  C[M,N](f32) = A[M,K](f16) * W[N,K]^T(f16) + bias[N]
// One wave per 64x32 macro-tile: 4 M-tiles x 2 N-tiles = 8 WMMA per K-block,
// 12 b128 loads per K-block (1.5 loads/WMMA).  K is a compile-time constant so
// every tile access is base + immediate offset (no VALU address arithmetic in
// the hot loop -> no WMMA->VALU hazard NOPs).  K%32==0, M%64==0, N%32==0.
// A/B fragments use contiguous k (k = 16*(lane/16) + s, not the ISA table order); f32 C/D per ISA 7.12.2.
// ---------------------------------------------------------------------------
template <int K>
__global__ __launch_bounds__(32)
void k_gemm_wmma(const _Float16* __restrict__ A,
                 const _Float16* __restrict__ W,
                 const float*    __restrict__ bias,
                 float*          __restrict__ C,
                 _Float16*       __restrict__ C16,
                 int N)
{
  const int lane = threadIdx.x;
  const int n0   = blockIdx.x << 5;            // 2 N-tiles of 16
  const int m0   = blockIdx.y << 6;            // 4 M-tiles of 16
  const int l16  = lane & 15;
  const int kh   = lane >> 4;                  // 0 | 1

  const _Float16* Ar = A + (size_t)(m0 + l16) * K + kh * 16;
  const _Float16* Wr = W + (size_t)(n0 + l16) * K + kh * 16;
  constexpr int As = 16 * K;                   // M-tile row stride (halves)
  constexpr int Ws = 16 * K;                   // N-tile row stride (halves)

  v8f acc[4][2] = {};                          // zero-init accumulators

  for (int k = 0; k < K; k += 32) {
    __builtin_prefetch((const void*)(Wr + k + 256), 0, 0);       // global_prefetch_b8
    __builtin_prefetch((const void*)(Wr + Ws + k + 256), 0, 0);

    v16h a[4], b[2];
#pragma unroll
    for (int i = 0; i < 4; ++i) {
      v8h a0 = *(const v8h*)(Ar + i * As + k);        // K = kb..kb+7
      v8h a1 = *(const v8h*)(Ar + i * As + k + 8);    // K = kb+8..kb+15
#pragma unroll
      for (int e = 0; e < 8; ++e) { a[i][e] = a0[e]; a[i][e + 8] = a1[e]; }
    }
#pragma unroll
    for (int j = 0; j < 2; ++j)
      b[j] = *(const v16h*)(Wr + j * Ws + k);         // 16 contiguous K

#pragma unroll
    for (int i = 0; i < 4; ++i)
#pragma unroll
      for (int j = 0; j < 2; ++j)
        acc[i][j] = __builtin_amdgcn_wmma_f32_16x16x32_f16(
            false, a[i], false, b[j], (short)0, acc[i][j], false, false);
  }

  const int mb = m0 + kh * 8;
#pragma unroll
  for (int j = 0; j < 2; ++j) {
    const int   n  = n0 + j * 16 + l16;
    const float bv = bias ? bias[n] : 0.f;
#pragma unroll
    for (int i = 0; i < 4; ++i) {
#pragma unroll
      for (int r = 0; r < 8; ++r) {
        float v = acc[i][j][r] + bv;
        size_t idx = (size_t)(mb + i * 16 + r) * N + n;
        if (C)   C[idx]   = v;
        if (C16) C16[idx] = (_Float16)v;
      }
    }
  }
}

// ---------------------------------------------------------------------------
// LSTM pointwise:  gates = gx + gh;  c' = sig(f)*c + sig(i)*tanh(g);
//                  h = sig(o)*tanh(c')
// ---------------------------------------------------------------------------
__global__ void k_lstm_step(const float* __restrict__ gx, int gxs,
                            const float* __restrict__ gh,
                            const float* __restrict__ cin,
                            float* __restrict__ cout,
                            float* __restrict__ csc, int cscs,
                            float* __restrict__ hsc, int hscs,
                            _Float16* __restrict__ h16,
                            int Hd, int n)
{
  int i = blockIdx.x * blockDim.x + threadIdx.x;
  if (i >= n) return;
  int b = i / Hd, j = i - b * Hd;
  const float* gxr = gx + (size_t)b * gxs;
  const float* ghr = gh + (size_t)b * 4 * Hd;
  float gi = gxr[j]          + ghr[j];
  float gf = gxr[Hd + j]     + ghr[Hd + j];
  float gg = gxr[2 * Hd + j] + ghr[2 * Hd + j];
  float go = gxr[3 * Hd + j] + ghr[3 * Hd + j];
  float c  = sigm(gf) * cin[i] + sigm(gi) * tanhf(gg);
  float h  = sigm(go) * tanhf(c);
  cout[i] = c;
  if (csc) csc[(size_t)b * cscs + j] = c;
  hsc[(size_t)b * hscs + j] = h;
  h16[i] = (_Float16)h;
}

// final = concat(Henc[b,last], Cenc[b,last])  -> f16 [B, 4H]
__global__ void k_final_gather(const float* __restrict__ Henc, const float* __restrict__ Cenc,
                               const int* __restrict__ src_lens, _Float16* __restrict__ out)
{
  int i = blockIdx.x * blockDim.x + threadIdx.x;
  if (i >= B_ * 2 * DH_) return;
  int b = i / (2 * DH_), j = i - b * (2 * DH_);
  int last = src_lens[b] - 1;
  size_t base = ((size_t)b * TS_ + last) * DH_;
  out[i] = (_Float16)((j < DH_) ? Henc[base + j] : Cenc[base + j - DH_]);
}

// h0 = tanh(c0), plus f16 mirror
__global__ void k_tanh_pair(const float* __restrict__ c, float* __restrict__ h,
                            _Float16* __restrict__ h16, int n)
{
  int i = blockIdx.x * blockDim.x + threadIdx.x;
  if (i >= n) return;
  float v = tanhf(c[i]);
  h[i] = v;
  h16[i] = (_Float16)v;
}

// decoder input v = concat(trg_emb[tok_s], av)  -> f16 [B, E+DH]
__global__ void k_dec_input(const int* __restrict__ trg, int s, const float* __restrict__ emb,
                            const _Float16* __restrict__ av16, _Float16* __restrict__ v)
{
  int i = blockIdx.x * blockDim.x + threadIdx.x;
  if (i >= B_ * (E_ + DH_)) return;
  int b = i / (E_ + DH_), j = i - b * (E_ + DH_);
  if (j < E_) v[i] = (_Float16)emb[(size_t)trg[b * TT_ + s] * E_ + j];
  else        v[i] = av16[b * DH_ + (j - E_)];
}

// scores[b,t] = <h[b], Henc[b,t]>, masked by src_lens
__global__ void k_attn_scores(const float* __restrict__ h, const float* __restrict__ Henc,
                              const int* __restrict__ src_lens, float* __restrict__ sc)
{
  int i = blockIdx.x * blockDim.x + threadIdx.x;
  if (i >= B_ * TS_) return;
  int b = i / TS_, t = i - b * TS_;
  const float* hr = h + (size_t)b * DH_;
  const float* er = Henc + ((size_t)b * TS_ + t) * DH_;
  float s = 0.f;
  for (int k = 0; k < DH_; ++k) s += hr[k] * er[k];
  sc[i] = (t < src_lens[b]) ? s : -1e9f;
}

// row softmax over TS=64 (one block of 64 per batch row)
__global__ void k_softmax_rows(const float* __restrict__ s, float* __restrict__ a)
{
  __shared__ float red[TS_];
  int b = blockIdx.x, t = threadIdx.x;
  float v = s[b * TS_ + t];
  red[t] = v; __syncthreads();
  for (int o = TS_ / 2; o > 0; o >>= 1) { if (t < o) red[t] = fmaxf(red[t], red[t + o]); __syncthreads(); }
  float mx = red[0]; __syncthreads();
  float e = __expf(v - mx);
  red[t] = e; __syncthreads();
  for (int o = TS_ / 2; o > 0; o >>= 1) { if (t < o) red[t] += red[t + o]; __syncthreads(); }
  a[b * TS_ + t] = e / red[0];
}

// ctx[b,j] = sum_t attn[b,t] * Henc[b,t,j]
__global__ void k_attn_ctx(const float* __restrict__ attn, const float* __restrict__ Henc,
                           float* __restrict__ ctx)
{
  int i = blockIdx.x * blockDim.x + threadIdx.x;
  if (i >= B_ * DH_) return;
  int b = i / DH_, j = i - b * DH_;
  float s = 0.f;
  for (int t = 0; t < TS_; ++t) s += attn[b * TS_ + t] * Henc[((size_t)b * TS_ + t) * DH_ + j];
  ctx[i] = s;
}

// hc = concat(h, ctx) -> f16 [B, 2DH]
__global__ void k_concat_f16(const float* __restrict__ h, const float* __restrict__ ctx,
                             _Float16* __restrict__ out)
{
  int i = blockIdx.x * blockDim.x + threadIdx.x;
  if (i >= B_ * 2 * DH_) return;
  int b = i / (2 * DH_), j = i - b * (2 * DH_);
  out[i] = (_Float16)((j < DH_) ? h[b * DH_ + j] : ctx[b * DH_ + j - DH_]);
}

// per-row log-softmax over VT; accumulate +lsm[gold] for masked positions
__global__ __launch_bounds__(256)
void k_nll(const float* __restrict__ logits, const int* __restrict__ trg_tokens,
           const int* __restrict__ trg_lens, int s, float* __restrict__ out)
{
  __shared__ float red[256];
  int b = blockIdx.x, t = threadIdx.x;
  const float* row = logits + (size_t)b * VT_;
  float mx = -1e30f;
  for (int i = t; i < VT_; i += 256) mx = fmaxf(mx, row[i]);
  red[t] = mx; __syncthreads();
  for (int o = 128; o > 0; o >>= 1) { if (t < o) red[t] = fmaxf(red[t], red[t + o]); __syncthreads(); }
  mx = red[0]; __syncthreads();
  float sum = 0.f;
  for (int i = t; i < VT_; i += 256) sum += __expf(row[i] - mx);
  red[t] = sum; __syncthreads();
  for (int o = 128; o > 0; o >>= 1) { if (t < o) red[t] += red[t + o]; __syncthreads(); }
  if (t == 0) {
    int gold = trg_tokens[b * TT_ + s + 1];
    float lsm = row[gold] - mx - __logf(red[0]);
    if (s + 1 < trg_lens[b]) atomicAdd(out, lsm);   // reference returns sum(lsm*mask)
  }
}

// ---------------------------------------------------------------------------
// Host side
// ---------------------------------------------------------------------------
static inline void gemm(hipStream_t st, const _Float16* A, const _Float16* W,
                        const float* bias, float* C, _Float16* C16, int M, int N, int K)
{
  dim3 g(N / 32, M / 64);                     // 64x32 macro-tile per wave
  switch (K) {
    case 512:  k_gemm_wmma<512> <<<g, 32, 0, st>>>(A, W, bias, C, C16, N); break;
    case 1024: k_gemm_wmma<1024><<<g, 32, 0, st>>>(A, W, bias, C, C16, N); break;
    case 1536: k_gemm_wmma<1536><<<g, 32, 0, st>>>(A, W, bias, C, C16, N); break;
    case 2048: k_gemm_wmma<2048><<<g, 32, 0, st>>>(A, W, bias, C, C16, N); break;
    default:   break;  // unused: all model GEMMs have K in {512,1024,1536,2048}
  }
}

extern "C" void kernel_launch(void* const* d_in, const int* in_sizes, int n_in,
                              void* d_out, int out_size, void* d_ws, size_t ws_size,
                              hipStream_t stream) {
  (void)in_sizes; (void)n_in; (void)out_size; (void)ws_size;

  const int*   src_tokens = (const int*)  d_in[0];
  const int*   src_lens   = (const int*)  d_in[1];
  const int*   trg_tokens = (const int*)  d_in[2];
  const int*   trg_lens   = (const int*)  d_in[3];
  const float* src_emb    = (const float*)d_in[4];
  const float* trg_emb    = (const float*)d_in[5];
  const float* enc_Wih    = (const float*)d_in[6];
  const float* enc_Whh    = (const float*)d_in[7];
  const float* enc_b      = (const float*)d_in[8];
  const float* rev_Wih    = (const float*)d_in[9];
  const float* rev_Whh    = (const float*)d_in[10];
  const float* rev_b      = (const float*)d_in[11];
  const float* dec_Wih    = (const float*)d_in[12];
  const float* dec_Whh    = (const float*)d_in[13];
  const float* dec_b      = (const float*)d_in[14];
  const float* hid_W      = (const float*)d_in[15];
  const float* hid_b      = (const float*)d_in[16];
  const float* out_W      = (const float*)d_in[17];
  const float* out_b      = (const float*)d_in[18];
  const float* init_W     = (const float*)d_in[19];
  const float* init_b     = (const float*)d_in[20];
  float* out = (float*)d_out;

  // ---- workspace bump allocator (~190 MB total) ----
  char* ws = (char*)d_ws;
  size_t off = 0;
  auto alloc = [&](size_t bytes) -> void* {
    void* p = ws + off;
    off += (bytes + 255) & ~(size_t)255;
    return p;
  };
  _Float16* enc_Wih16 = (_Float16*)alloc((size_t)FH_ * E_ * 2);
  _Float16* enc_Whh16 = (_Float16*)alloc((size_t)FH_ * H_ * 2);
  _Float16* rev_Wih16 = (_Float16*)alloc((size_t)FH_ * E_ * 2);
  _Float16* rev_Whh16 = (_Float16*)alloc((size_t)FH_ * H_ * 2);
  _Float16* dec_Wih16 = (_Float16*)alloc((size_t)FDH_ * (E_ + DH_) * 2);
  _Float16* dec_Whh16 = (_Float16*)alloc((size_t)FDH_ * DH_ * 2);
  _Float16* hid_W16   = (_Float16*)alloc((size_t)DH_ * 2 * DH_ * 2);
  _Float16* out_W16   = (_Float16*)alloc((size_t)VT_ * DH_ * 2);
  _Float16* init_W16  = (_Float16*)alloc((size_t)DH_ * 2 * DH_ * 2);
  _Float16* sv16      = (_Float16*)alloc((size_t)B_ * TS_ * E_ * 2);
  float*    xg        = (float*)   alloc((size_t)B_ * TS_ * FH_ * 4);   // reused per dir
  float*    Henc      = (float*)   alloc((size_t)B_ * TS_ * DH_ * 4);   // [B,TS,2H]
  float*    Cenc      = (float*)   alloc((size_t)B_ * TS_ * DH_ * 4);
  _Float16* h16e      = (_Float16*)alloc((size_t)B_ * H_ * 2);
  float*    ce        = (float*)   alloc((size_t)B_ * H_ * 4);
  float*    ghe       = (float*)   alloc((size_t)B_ * FH_ * 4);
  _Float16* final16   = (_Float16*)alloc((size_t)B_ * 2 * DH_ * 2);
  float*    cd        = (float*)   alloc((size_t)B_ * DH_ * 4);
  float*    hd        = (float*)   alloc((size_t)B_ * DH_ * 4);
  _Float16* h16d      = (_Float16*)alloc((size_t)B_ * DH_ * 2);
  _Float16* av16      = (_Float16*)alloc((size_t)B_ * DH_ * 2);
  _Float16* vin16     = (_Float16*)alloc((size_t)B_ * (E_ + DH_) * 2);
  float*    gxd       = (float*)   alloc((size_t)B_ * FDH_ * 4);
  float*    ghd       = (float*)   alloc((size_t)B_ * FDH_ * 4);
  float*    scores    = (float*)   alloc((size_t)B_ * TS_ * 4);
  float*    attn      = (float*)   alloc((size_t)B_ * TS_ * 4);
  float*    ctx       = (float*)   alloc((size_t)B_ * DH_ * 4);
  _Float16* hc16      = (_Float16*)alloc((size_t)B_ * 2 * DH_ * 2);
  float*    logits    = (float*)   alloc((size_t)B_ * VT_ * 4);

  auto cvt = [&](const float* s, _Float16* d, long n) {
    k_f32_to_f16<<<(unsigned)((n + 255) / 256), 256, 0, stream>>>(s, d, n);
  };

  // ---- output accumulator + weight conversion (f32 -> f16, once per launch) ----
  k_zero_f32<<<1, 32, 0, stream>>>(out, 1);
  cvt(enc_Wih, enc_Wih16, (long)FH_ * E_);
  cvt(enc_Whh, enc_Whh16, (long)FH_ * H_);
  cvt(rev_Wih, rev_Wih16, (long)FH_ * E_);
  cvt(rev_Whh, rev_Whh16, (long)FH_ * H_);
  cvt(dec_Wih, dec_Wih16, (long)FDH_ * (E_ + DH_));
  cvt(dec_Whh, dec_Whh16, (long)FDH_ * DH_);
  cvt(hid_W,   hid_W16,   (long)DH_ * 2 * DH_);
  cvt(out_W,   out_W16,   (long)VT_ * DH_);   // 65.5 MB f16 -> resident in 192 MB L2
  cvt(init_W,  init_W16,  (long)DH_ * 2 * DH_);

  // ---- encode: gather embeddings, then two directions (reference runs both forward) ----
  {
    long n = (long)B_ * TS_ * E_;
    k_gather_rows_f16<<<(unsigned)((n + 255) / 256), 256, 0, stream>>>(src_tokens, src_emb, sv16, E_, n);
  }
  struct Dir { const _Float16 *Wih, *Whh; const float* b; int doff; };
  Dir dirs[2] = { {enc_Wih16, enc_Whh16, enc_b, 0}, {rev_Wih16, rev_Whh16, rev_b, H_} };
  for (int d = 0; d < 2; ++d) {
    // all-timestep input GEMM: [B*TS, E] x [E, 4H]
    gemm(stream, sv16, dirs[d].Wih, dirs[d].b, xg, nullptr, B_ * TS_, FH_, E_);
    k_zero_f16<<<(B_ * H_ + 255) / 256, 256, 0, stream>>>(h16e, B_ * H_);
    k_zero_f32<<<(B_ * H_ + 255) / 256, 256, 0, stream>>>(ce, B_ * H_);
    for (int t = 0; t < TS_; ++t) {
      gemm(stream, h16e, dirs[d].Whh, nullptr, ghe, nullptr, B_, FH_, H_);
      k_lstm_step<<<(B_ * H_ + 255) / 256, 256, 0, stream>>>(
          xg + (size_t)t * FH_, TS_ * FH_, ghe, ce, ce,
          Cenc + (size_t)t * DH_ + dirs[d].doff, TS_ * DH_,
          Henc + (size_t)t * DH_ + dirs[d].doff, TS_ * DH_,
          h16e, H_, B_ * H_);
    }
  }

  // ---- decoder init: c0 = final @ init_W^T + b ; h0 = tanh(c0) ; av0 = 0 ----
  k_final_gather<<<(B_ * 2 * DH_ + 255) / 256, 256, 0, stream>>>(Henc, Cenc, src_lens, final16);
  gemm(stream, final16, init_W16, init_b, cd, nullptr, B_, DH_, 2 * DH_);
  k_tanh_pair<<<(B_ * DH_ + 255) / 256, 256, 0, stream>>>(cd, hd, h16d, B_ * DH_);
  k_zero_f16<<<(B_ * DH_ + 255) / 256, 256, 0, stream>>>(av16, B_ * DH_);

  // ---- decode: 63 teacher-forced steps ----
  for (int s = 0; s < TT_ - 1; ++s) {
    k_dec_input<<<(B_ * (E_ + DH_) + 255) / 256, 256, 0, stream>>>(trg_tokens, s, trg_emb, av16, vin16);
    gemm(stream, vin16, dec_Wih16, dec_b, gxd, nullptr, B_, FDH_, E_ + DH_);
    gemm(stream, h16d,  dec_Whh16, nullptr, ghd, nullptr, B_, FDH_, DH_);
    k_lstm_step<<<(B_ * DH_ + 255) / 256, 256, 0, stream>>>(
        gxd, FDH_, ghd, cd, cd, nullptr, 0, hd, DH_, h16d, DH_, B_ * DH_);
    k_attn_scores<<<(B_ * TS_ + 255) / 256, 256, 0, stream>>>(hd, Henc, src_lens, scores);
    k_softmax_rows<<<B_, TS_, 0, stream>>>(scores, attn);
    k_attn_ctx<<<(B_ * DH_ + 255) / 256, 256, 0, stream>>>(attn, Henc, ctx);
    k_concat_f16<<<(B_ * 2 * DH_ + 255) / 256, 256, 0, stream>>>(hd, ctx, hc16);
    gemm(stream, hc16, hid_W16, hid_b, nullptr, av16, B_, DH_, 2 * DH_);       // av (f16 only)
    gemm(stream, av16, out_W16, out_b, logits, nullptr, B_, VT_, DH_);         // dominant GEMM
    k_nll<<<B_, 256, 0, stream>>>(logits, trg_tokens, trg_lens, s, out);
  }
}
